// GNNDrugEncoder_6399501271497
// MI455X (gfx1250) — matrix operand order repaired and checked by reading, not compiled, not run on hardware
//
#include <hip/hip_runtime.h>
#include <hip/hip_bf16.h>
#include <math.h>

// ---------------------------------------------------------------------------
// Types for CDNA5 WMMA (wave32, 16x16x32 bf16 -> f32)
// ---------------------------------------------------------------------------
typedef __attribute__((ext_vector_type(8)))  __bf16 v8bf;
typedef __attribute__((ext_vector_type(16))) __bf16 v16bf;
typedef __attribute__((ext_vector_type(8)))  float  v8f;

#define HID   256
#define NL    4
#define NBOND 6
#define NATOM 38
#define OUTD  256
#define BB    32
#define NN    128
#define MTOT  (BB * NN)   // 4096 rows
#define KPAD  32          // bond K padded 6 -> 32 for WMMA

// round-to-nearest-even fp32 -> bf16
__device__ __forceinline__ unsigned short f2bf(float f) {
  unsigned int u = __float_as_uint(f);
  unsigned int r = u + 0x7FFFu + ((u >> 16) & 1u);
  return (unsigned short)(r >> 16);
}

__device__ __forceinline__ v16bf combine16(v8bf lo, v8bf hi) {
  return __builtin_shufflevector(lo, hi, 0,1,2,3,4,5,6,7,8,9,10,11,12,13,14,15);
}

// ---------------------------------------------------------------------------
// Weight conversion:
//  - msg_w[:, :H, :] transposed to (l, n, k) bf16  (Bt layout for WMMA)
//  - w_ih / w_hh (3H, H): already Bt layout, straight convert
//  - Wb = msg_w[:, H:H+6, :] -> (l, n, kpad32) bf16, zero-padded K
// ---------------------------------------------------------------------------
__global__ void convert_weights_kernel(const float* __restrict__ msgw,
                                       const float* __restrict__ wih,
                                       const float* __restrict__ whh,
                                       unsigned short* __restrict__ whT,
                                       unsigned short* __restrict__ wihb,
                                       unsigned short* __restrict__ whhb,
                                       unsigned short* __restrict__ wbPad) {
  const int n0 = NL * HID * HID;        // 262144
  const int n1 = NL * 3 * HID * HID;    // 786432
  const int n2 = NL * HID * KPAD;       // 32768
  int idx = blockIdx.x * blockDim.x + threadIdx.x;
  if (idx < n0) {
    int l = idx >> 16, rem = idx & 65535;
    int n = rem >> 8, k = rem & 255;
    whT[idx] = f2bf(msgw[l * (HID + NBOND) * HID + k * HID + n]);
  } else if (idx < n0 + n1) {
    int j = idx - n0;
    wihb[j] = f2bf(wih[j]);
  } else if (idx < n0 + 2 * n1) {
    int j = idx - n0 - n1;
    whhb[j] = f2bf(whh[j]);
  } else if (idx < n0 + 2 * n1 + n2) {
    int j = idx - n0 - 2 * n1;
    int l = j / (HID * KPAD), rem = j % (HID * KPAD);
    int n = rem >> 5, k = rem & 31;
    wbPad[j] = (k < NBOND)
        ? f2bf(msgw[l * (HID + NBOND) * HID + (HID + k) * HID + n])
        : (unsigned short)0;
  }
}

// ---------------------------------------------------------------------------
// Embed: h = relu(node @ embed_w + embed_b) * mask   (K = 38, VALU)
// ---------------------------------------------------------------------------
__global__ __launch_bounds__(HID) void embed_kernel(
    const float* __restrict__ node, const float* __restrict__ embw,
    const float* __restrict__ embb, const float* __restrict__ mask,
    float* __restrict__ h, unsigned short* __restrict__ h_bf) {
  __shared__ float s_node[NATOM];
  const int m = blockIdx.x, c = threadIdx.x;
  if (c < NATOM) s_node[c] = node[(size_t)m * NATOM + c];
  __syncthreads();
  float acc = embb[c];
#pragma unroll
  for (int k = 0; k < NATOM; ++k)
    acc = fmaf(s_node[k], embw[k * HID + c], acc);
  float v = fmaxf(acc, 0.f) * mask[m];
  h[(size_t)m * HID + c] = v;
  h_bf[(size_t)m * HID + c] = f2bf(v);
}

// ---------------------------------------------------------------------------
// WMMA GEMM:  C[m][n] = sum_k A[m][k] * Bt[n][k]  (+ bias[n])
// Block = 128 threads (4 waves); wave computes 16 rows x 64 cols.
// Grid: x = M/64, y = N/64.  K multiple of 32.
// ---------------------------------------------------------------------------
__global__ __launch_bounds__(128) void gemm_bf16_kernel(
    const __bf16* __restrict__ A, const __bf16* __restrict__ Bt,
    const float* __restrict__ bias, float* __restrict__ C,
    int M, int N, int K) {
  const int lane = threadIdx.x & 31;
  const int wid  = threadIdx.x >> 5;
  const int m0   = (blockIdx.x * 4 + wid) * 16;
  const int n0   = blockIdx.y * 64;
  const int l16  = lane & 15;
  const int kh   = lane >> 4;

  const __bf16* arow  = A  + (size_t)(m0 + l16) * K;
  const __bf16* brow0 = Bt + (size_t)(n0 +  0 + l16) * K + kh * 16;
  const __bf16* brow1 = Bt + (size_t)(n0 + 16 + l16) * K + kh * 16;
  const __bf16* brow2 = Bt + (size_t)(n0 + 32 + l16) * K + kh * 16;
  const __bf16* brow3 = Bt + (size_t)(n0 + 48 + l16) * K + kh * 16;

  v8f c0 = {}, c1 = {}, c2 = {}, c3 = {};

  for (int k0 = 0; k0 < K; k0 += 32) {
    v16bf a = combine16(*(const v8bf*)(arow + k0 + kh * 16),
                        *(const v8bf*)(arow + k0 + kh * 16 + 8));
    v16bf b0 = combine16(*(const v8bf*)(brow0 + k0), *(const v8bf*)(brow0 + k0 + 8));
    v16bf b1 = combine16(*(const v8bf*)(brow1 + k0), *(const v8bf*)(brow1 + k0 + 8));
    v16bf b2 = combine16(*(const v8bf*)(brow2 + k0), *(const v8bf*)(brow2 + k0 + 8));
    v16bf b3 = combine16(*(const v8bf*)(brow3 + k0), *(const v8bf*)(brow3 + k0 + 8));
    c0 = __builtin_amdgcn_wmma_f32_16x16x32_bf16(false, a, false, b0, (short)0, c0, false, false);
    c1 = __builtin_amdgcn_wmma_f32_16x16x32_bf16(false, a, false, b1, (short)0, c1, false, false);
    c2 = __builtin_amdgcn_wmma_f32_16x16x32_bf16(false, a, false, b2, (short)0, c2, false, false);
    c3 = __builtin_amdgcn_wmma_f32_16x16x32_bf16(false, a, false, b3, (short)0, c3, false, false);
  }

  const int mrow = m0 + kh * 8;
#pragma unroll
  for (int t = 0; t < 4; ++t) {
    const int col = n0 + t * 16 + l16;
    const float bv = bias ? bias[col] : 0.f;
    const v8f acc = (t == 0) ? c0 : (t == 1) ? c1 : (t == 2) ? c2 : c3;
#pragma unroll
    for (int r = 0; r < 8; ++r)
      C[(size_t)(mrow + r) * N + col] = acc[r] + bv;
  }
}

// ---------------------------------------------------------------------------
// Fused message + aggregation with WMMA for the bond@Wb product:
//   agg[m][h] = sum_j adj[m][j] * relu(hp[m][h] + (bond[m][j][:] @ Wb)[h])
// hp already contains msg_b (GEMM bias).
// One block (256 thr = 8 waves) per row m. Bond row staged in LDS as bf16
// padded to K=32; Wb pre-padded to (256 x 32) bf16 so each wave keeps its
// 2 B fragments in registers across the whole j loop. Only the relu +
// adj-weighted reduction (un-factorable through the relu) stays on VALU.
// ---------------------------------------------------------------------------
__global__ __launch_bounds__(256) void msg_agg_wmma_kernel(
    const float* __restrict__ hp, const float* __restrict__ bond,
    const float* __restrict__ adj, const __bf16* __restrict__ wbPad,
    unsigned short* __restrict__ agg_bf) {
  __shared__ unsigned short s_bond[NN * KPAD];   // 128x32 bf16 = 8 KB
  __shared__ float s_adj[NN];
  const int m = blockIdx.x;
  const int tid = threadIdx.x;
  const int lane = tid & 31, w = tid >> 5;
  const int l16 = lane & 15, kh = lane >> 4;

  // stage bond row -> bf16, zero-padded to K=32
  const float* brow = bond + (size_t)m * NN * NBOND;
  for (int t = tid; t < NN * KPAD; t += 256) {
    int j = t >> 5, k = t & 31;
    s_bond[t] = (k < NBOND) ? f2bf(brow[j * NBOND + k]) : (unsigned short)0;
  }
  if (tid < NN) s_adj[tid] = adj[(size_t)m * NN + tid];
  __syncthreads();

  // wave w owns h columns [32w, 32w+32): two 16-wide tiles, B frags loaded once
  const int h0 = w * 32;
  const __bf16* bp0 = wbPad + (size_t)(h0 +      l16) * KPAD + kh * 16;
  const __bf16* bp1 = wbPad + (size_t)(h0 + 16 + l16) * KPAD + kh * 16;
  const v16bf b0 = combine16(*(const v8bf*)bp0, *(const v8bf*)(bp0 + 8));
  const v16bf b1 = combine16(*(const v8bf*)bp1, *(const v8bf*)(bp1 + 8));

  const float hp0 = hp[(size_t)m * HID + h0 + l16];
  const float hp1 = hp[(size_t)m * HID + h0 + 16 + l16];

  float acc0 = 0.f, acc1 = 0.f;
#pragma unroll
  for (int j0 = 0; j0 < NN; j0 += 16) {
    const __bf16* ap = (const __bf16*)s_bond + (j0 + l16) * KPAD + kh * 16;
    v16bf a = combine16(*(const v8bf*)ap, *(const v8bf*)(ap + 8));
    v8f d0 = {}, d1 = {};
    d0 = __builtin_amdgcn_wmma_f32_16x16x32_bf16(false, a, false, b0, (short)0, d0, false, false);
    d1 = __builtin_amdgcn_wmma_f32_16x16x32_bf16(false, a, false, b1, (short)0, d1, false, false);
    // D layout: lanes 0-15 -> j = j0+r; lanes 16-31 -> j = j0+8+r (h = n col)
    const int jb = j0 + kh * 8;
#pragma unroll
    for (int r = 0; r < 8; ++r) {
      float aj = s_adj[jb + r];                       // LDS broadcast
      acc0 = fmaf(aj, fmaxf(hp0 + d0[r], 0.f), acc0);
      acc1 = fmaf(aj, fmaxf(hp1 + d1[r], 0.f), acc1);
    }
  }
  // lanes L and L+16 hold partials for the same h column: combine halves
  acc0 += __shfl_xor(acc0, 16, 32);
  acc1 += __shfl_xor(acc1, 16, 32);
  if (lane < 16) {
    agg_bf[(size_t)m * HID + h0 + lane]      = f2bf(acc0);
    agg_bf[(size_t)m * HID + h0 + 16 + lane] = f2bf(acc1);
  }
}

// ---------------------------------------------------------------------------
// GRU gates + LayerNorm + mask. One block per row m; 256 threads = channels.
// ---------------------------------------------------------------------------
__global__ __launch_bounds__(HID) void gru_ln_kernel(
    const float* __restrict__ gx, const float* __restrict__ gh,
    const float* __restrict__ lng, const float* __restrict__ lnb,
    const float* __restrict__ mask, float* __restrict__ h,
    unsigned short* __restrict__ h_bf) {
  __shared__ float sred[HID];
  const int m = blockIdx.x, c = threadIdx.x;
  const size_t g = (size_t)m * 3 * HID;

  float xr = gx[g + c], xz = gx[g + HID + c], xn = gx[g + 2 * HID + c];
  float hr = gh[g + c], hz = gh[g + HID + c], hn = gh[g + 2 * HID + c];
  float r = 1.f / (1.f + expf(-(xr + hr)));
  float z = 1.f / (1.f + expf(-(xz + hz)));
  float n = tanhf(xn + r * hn);
  float hv = h[(size_t)m * HID + c];
  float hnew = (1.f - z) * n + z * hv;

  sred[c] = hnew; __syncthreads();
  for (int s = HID / 2; s > 0; s >>= 1) {
    if (c < s) sred[c] += sred[c + s];
    __syncthreads();
  }
  float mu = sred[0] * (1.f / HID);
  __syncthreads();
  float d = hnew - mu;
  sred[c] = d * d; __syncthreads();
  for (int s = HID / 2; s > 0; s >>= 1) {
    if (c < s) sred[c] += sred[c + s];
    __syncthreads();
  }
  float var = sred[0] * (1.f / HID);

  float y = d * rsqrtf(var + 1e-5f) * lng[c] + lnb[c];
  y *= mask[m];
  h[(size_t)m * HID + c] = y;
  h_bf[(size_t)m * HID + c] = f2bf(y);
}

// ---------------------------------------------------------------------------
// Readout: masked mean + masked max over N. Block per batch b.
// ---------------------------------------------------------------------------
__global__ __launch_bounds__(HID) void readout_kernel(
    const float* __restrict__ h, const float* __restrict__ mask,
    float* __restrict__ pooled) {
  const int b = blockIdx.x, c = threadIdx.x;
  float sum = 0.f, msum = 0.f, mx = -1000000000.0f;
  for (int i = 0; i < NN; ++i) {
    const int m = b * NN + i;
    float me = mask[m];
    float v = h[(size_t)m * HID + c];
    sum = fmaf(v, me, sum);
    msum += me;
    float cand = (me == 0.f) ? -1000000000.0f : v;
    mx = fmaxf(mx, cand);
  }
  pooled[b * (2 * HID) + c] = sum / fmaxf(msum, 1e-6f);
  pooled[b * (2 * HID) + HID + c] = mx;
}

// ---------------------------------------------------------------------------
// Projections (tiny GEMVs: M=32). Exact-erf GELU to match reference.
// ---------------------------------------------------------------------------
__global__ __launch_bounds__(HID) void proj1_kernel(
    const float* __restrict__ pooled, const float* __restrict__ w1,
    const float* __restrict__ b1, float* __restrict__ hid) {
  __shared__ float sp[2 * HID];
  const int b = blockIdx.x, c = threadIdx.x;
  sp[c] = pooled[b * (2 * HID) + c];
  sp[c + HID] = pooled[b * (2 * HID) + HID + c];
  __syncthreads();
  float acc = b1[c];
  for (int k = 0; k < 2 * HID; ++k)
    acc = fmaf(sp[k], w1[k * HID + c], acc);
  float g = 0.5f * acc * (1.f + erff(acc * 0.70710678118654752f));
  hid[b * HID + c] = g;
}

__global__ __launch_bounds__(OUTD) void proj2_kernel(
    const float* __restrict__ hid, const float* __restrict__ w2,
    const float* __restrict__ b2, float* __restrict__ out) {
  __shared__ float sp[HID];
  const int b = blockIdx.x, c = threadIdx.x;
  sp[c] = hid[b * HID + c];
  __syncthreads();
  float acc = b2[c];
  for (int k = 0; k < HID; ++k)
    acc = fmaf(sp[k], w2[k * OUTD + c], acc);
  out[b * OUTD + c] = acc;
}

// ---------------------------------------------------------------------------
// Host launch
// ---------------------------------------------------------------------------
extern "C" void kernel_launch(void* const* d_in, const int* in_sizes, int n_in,
                              void* d_out, int out_size, void* d_ws, size_t ws_size,
                              hipStream_t stream) {
  (void)in_sizes; (void)n_in; (void)out_size; (void)ws_size;
  const float* node = (const float*)d_in[0];
  const float* adj  = (const float*)d_in[1];
  const float* bond = (const float*)d_in[2];
  const float* mask = (const float*)d_in[3];
  const float* embw = (const float*)d_in[4];
  const float* embb = (const float*)d_in[5];
  const float* msgw = (const float*)d_in[6];
  const float* msgb = (const float*)d_in[7];
  const float* wih  = (const float*)d_in[8];
  const float* whh  = (const float*)d_in[9];
  const float* bih  = (const float*)d_in[10];
  const float* bhh  = (const float*)d_in[11];
  const float* lng  = (const float*)d_in[12];
  const float* lnb  = (const float*)d_in[13];
  const float* pw1  = (const float*)d_in[14];
  const float* pb1  = (const float*)d_in[15];
  const float* pw2  = (const float*)d_in[16];
  const float* pb2  = (const float*)d_in[17];
  float* out = (float*)d_out;

  char* p = (char*)d_ws;
  auto carve = [&](size_t bytes) -> void* {
    void* r = (void*)p;
    p += (bytes + 255) & ~(size_t)255;
    return r;
  };
  float* h    = (float*)carve((size_t)MTOT * HID * 4);          // 4 MB
  float* hp   = (float*)carve((size_t)MTOT * HID * 4);          // 4 MB
  float* gx   = (float*)carve((size_t)MTOT * 3 * HID * 4);      // 12 MB
  float* gh   = (float*)carve((size_t)MTOT * 3 * HID * 4);      // 12 MB
  unsigned short* h_bf   = (unsigned short*)carve((size_t)MTOT * HID * 2);
  unsigned short* agg_bf = (unsigned short*)carve((size_t)MTOT * HID * 2);
  unsigned short* whT    = (unsigned short*)carve((size_t)NL * HID * HID * 2);
  unsigned short* wihb   = (unsigned short*)carve((size_t)NL * 3 * HID * HID * 2);
  unsigned short* whhb   = (unsigned short*)carve((size_t)NL * 3 * HID * HID * 2);
  unsigned short* wbPad  = (unsigned short*)carve((size_t)NL * HID * KPAD * 2);
  float* pooled = (float*)carve((size_t)BB * 2 * HID * 4);
  float* hid    = (float*)carve((size_t)BB * HID * 4);

  // 1) weights -> bf16 (whT transposed, Wb padded to K=32)
  {
    int total = NL * HID * HID + 2 * (NL * 3 * HID * HID) + NL * HID * KPAD;
    convert_weights_kernel<<<(total + 255) / 256, 256, 0, stream>>>(
        msgw, wih, whh, whT, wihb, whhb, wbPad);
  }

  // 2) embed
  embed_kernel<<<MTOT, HID, 0, stream>>>(node, embw, embb, mask, h, h_bf);

  // 3) layers
  for (int l = 0; l < NL; ++l) {
    const unsigned short* whT_l = whT   + (size_t)l * HID * HID;
    const unsigned short* wih_l = wihb  + (size_t)l * 3 * HID * HID;
    const unsigned short* whh_l = whhb  + (size_t)l * 3 * HID * HID;
    const unsigned short* wb_l  = wbPad + (size_t)l * HID * KPAD;
    const float* msgb_l = msgb + l * HID;
    const float* bih_l  = bih + l * 3 * HID;
    const float* bhh_l  = bhh + l * 3 * HID;
    const float* lng_l  = lng + l * HID;
    const float* lnb_l  = lnb + l * HID;

    // hp = h @ Wh + msg_b   (WMMA, M=4096 N=256 K=256)
    gemm_bf16_kernel<<<dim3(MTOT / 64, HID / 64), 128, 0, stream>>>(
        (const __bf16*)h_bf, (const __bf16*)whT_l, msgb_l, hp, MTOT, HID, HID);

    // fused message + aggregation (WMMA for bond@Wb, VALU for relu/agg)
    msg_agg_wmma_kernel<<<MTOT, 256, 0, stream>>>(
        hp, bond, adj, (const __bf16*)wb_l, agg_bf);

    // gx = agg @ w_ih^T + b_ih  (WMMA, N=768)
    gemm_bf16_kernel<<<dim3(MTOT / 64, (3 * HID) / 64), 128, 0, stream>>>(
        (const __bf16*)agg_bf, (const __bf16*)wih_l, bih_l, gx, MTOT, 3 * HID, HID);

    // gh = h @ w_hh^T + b_hh   (WMMA, N=768)
    gemm_bf16_kernel<<<dim3(MTOT / 64, (3 * HID) / 64), 128, 0, stream>>>(
        (const __bf16*)h_bf, (const __bf16*)whh_l, bhh_l, gh, MTOT, 3 * HID, HID);

    // GRU + LayerNorm + mask
    gru_ln_kernel<<<MTOT, HID, 0, stream>>>(gx, gh, lng_l, lnb_l, mask, h, h_bf);
  }

  // 4) readout + projections
  readout_kernel<<<BB, HID, 0, stream>>>(h, mask, pooled);
  proj1_kernel<<<BB, HID, 0, stream>>>(pooled, pw1, pb1, hid);
  proj2_kernel<<<BB, OUTD, 0, stream>>>(hid, pw2, pb2, out);
}
